// HyperbolicVortexLayer_7679401525691
// MI455X (gfx1250) — matrix operand order repaired and checked by reading, not compiled, not run on hardware
//
#include <hip/hip_runtime.h>

#define HIDDEN     512
#define NUM_NODES  9
#define BATCH      8192
#define BT         8                    // batch elements per block
#define ROWS       (BT * NUM_NODES)     // 72 matrix rows per block
#define ROWS_PAD   80                   // padded to 5 tiles of 16
#define THREADS    512
#define NWAVES     16
#define KSLAB      32
#define NSLABS     (HIDDEN / KSLAB)     // 16
#define SLAB_ELEMS (HIDDEN * KSLAB)     // 512*32 f16 per slab buffer
#define EPSF       1e-8f

typedef __attribute__((ext_vector_type(16))) _Float16 v16h;
typedef __attribute__((ext_vector_type(8)))  _Float16 v8h;
typedef __attribute__((ext_vector_type(4)))  _Float16 v4h;
typedef __attribute__((ext_vector_type(8)))  float    v8f;
typedef __attribute__((ext_vector_type(4)))  unsigned int v4u;
typedef __attribute__((ext_vector_type(8)))  int      v8i;
typedef __attribute__((ext_vector_type(4)))  int      v4i;

// ---------------- LDS layout (dynamic shared) ----------------
// sH : ROWS_PAD x 512 f32  (x_proj, then h)            160 KB
// sA : ROWS_PAD x 512 f16  (A staging, then result)     80 KB
// sB : 2 x (512 x 32) f16  (TDM-staged weight slabs)    64 KB
#define SH_OFF 0
#define SA_OFF 163840
#define SB_OFF 245760
#define SMEM_BYTES 311296

// neighbor lists replicating the reference's torch.nonzero scan order
__device__ const int g_off[10] = {0, 3, 6, 8, 11, 14, 16, 19, 22, 30};
__device__ const int g_nbr[30] = {4,7,8, 0,6,8, 5,8, 1,4,8, 6,3,8, 2,8,
                                  7,1,8, 3,0,8, 0,1,2,3,4,5,6,7};

__global__ void cvt_weights(const float* __restrict__ a, const float* __restrict__ b,
                            _Float16* __restrict__ oa, _Float16* __restrict__ ob) {
  const int N = HIDDEN * HIDDEN;
  int i = blockIdx.x * blockDim.x + threadIdx.x;
  if (i < N)            oa[i]     = (_Float16)a[i];
  else if (i < 2 * N)   ob[i - N] = (_Float16)b[i - N];
}

__device__ __forceinline__ float wave_sum(float v) {
#pragma unroll
  for (int off = 16; off > 0; off >>= 1) v += __shfl_xor(v, off, 32);
  return v;
}

// o = mobius_add(x, y, c); x,y are 16 f32 per lane (element l + 32t of 512)
__device__ __forceinline__ void mobius_add(const float (&x)[16], const float (&y)[16],
                                           float c, float (&o)[16]) {
  float xy = 0.f, xx = 0.f, yy = 0.f;
#pragma unroll
  for (int t = 0; t < 16; ++t) { xy += x[t]*y[t]; xx += x[t]*x[t]; yy += y[t]*y[t]; }
  xy = wave_sum(xy); xx = wave_sum(xx); yy = wave_sum(yy);
  const float ca  = 1.f + 2.f*c*xy + c*yy;
  const float cb  = 1.f - c*xx;
  const float inv = 1.f / (1.f + 2.f*c*xy + c*c*xx*yy + EPSF);
#pragma unroll
  for (int t = 0; t < 16; ++t) o[t] = (ca*x[t] + cb*y[t]) * inv;
}

// TDM: DMA a 512x32 f16 tile (columns [ks*32, ks*32+32) of a 512x512 f16
// matrix, row-major) into LDS, packed row-major as [n][32] -- exactly the
// B-slab layout the WMMA fragment loads expect.  D# per ISA ch.8.
__device__ __forceinline__ void tdm_load_slab(const _Float16* __restrict__ W,
                                              int ks, _Float16* ldsDst) {
  const unsigned long long ga = (unsigned long long)(const void*)(W + ks * KSLAB);
  const unsigned lds = (unsigned)(unsigned long long)(void*)ldsDst;  // addr[31:0] = LDS offset
  v4u g0;
  g0[0] = 1u;                                              // count=1, user D#, no gather
  g0[1] = lds;                                             // lds_addr (bytes)
  g0[2] = (unsigned)(ga & 0xFFFFFFFFu);                    // global_addr[31:0]
  g0[3] = (unsigned)((ga >> 32) & 0x01FFFFFFu) | (2u << 30); // global_addr[56:32] | type=2
  v8i g1;
  g1[0] = (int)(1u << 16);          // workgroup_mask=0, data_size=1 (2 bytes)
  g1[1] = (int)(512u << 16);        // tensor_dim0[15:0]=512   (bits 63:48)
  g1[2] = (int)(512u << 16);        // dim0 hi=0; tensor_dim1[15:0]=512 (bits 95:80)
  g1[3] = (int)((unsigned)KSLAB << 16); // dim1 hi=0; tile_dim0=32 (bits 127:112)
  g1[4] = 512;                      // tile_dim1=512, tile_dim2=0
  g1[5] = 512;                      // tensor_dim0_stride[31:0]=512
  g1[6] = 0;                        // stride hi / tensor_dim1_stride lo
  g1[7] = 0;
  const v4i gz = {0, 0, 0, 0};
#if defined(__clang_major__) && (__clang_major__ >= 23)
  const v8i gz8 = {0, 0, 0, 0, 0, 0, 0, 0};
  __builtin_amdgcn_tensor_load_to_lds(g0, g1, gz, gz, gz8, 0);
#else
  __builtin_amdgcn_tensor_load_to_lds(g0, g1, gz, gz, 0);
#endif
}

// C[80x512] += A[80x512] * W^T via v_wmma_f32_16x16x32_f16.
// B slabs are TDM-DMA'd into a double buffer; DMA of slab ks+1 overlaps
// WMMA work on slab ks. Each wave: 5 M-tiles x 2 N-tiles.
__device__ __forceinline__ void run_gemm(const _Float16* __restrict__ sA,
                                         _Float16* __restrict__ sB,
                                         const _Float16* __restrict__ W,
                                         int tid, v8f acc[5][2]) {
  const int lane = tid & 31;
  const int wave = tid >> 5;
  const int l16  = lane & 15;
  const int half = lane >> 4;
  const v8f vz = {0.f, 0.f, 0.f, 0.f, 0.f, 0.f, 0.f, 0.f};
#pragma unroll
  for (int mt = 0; mt < 5; ++mt) { acc[mt][0] = vz; acc[mt][1] = vz; }

  if (wave == 0) tdm_load_slab(W, 0, sB);            // prime buffer 0

  for (int ks = 0; ks < NSLABS; ++ks) {
    if (wave == 0) __builtin_amdgcn_s_wait_tensorcnt(0);  // slab ks landed
    __syncthreads();                                      // publish; prev slab consumed
    if (wave == 0 && ks + 1 < NSLABS)
      tdm_load_slab(W, ks + 1, sB + ((ks + 1) & 1) * SLAB_ELEMS);

    const _Float16* sBk = sB + (ks & 1) * SLAB_ELEMS;

    // B fragment: lane holds B[k = half*16 .. +15][n] = W[n][k...] contiguous
    v16h bfrag[2];
#pragma unroll
    for (int j = 0; j < 2; ++j) {
      const int n = wave * 32 + j * 16 + l16;
      bfrag[j] = *(const v16h*)(sBk + n * KSLAB + half * 16);
    }
#pragma unroll
    for (int mt = 0; mt < 5; ++mt) {
      // A fragment: row m = mt*16 + l16, K = half*16 .. +15 contiguous
      const int m = mt * 16 + l16;
      const v8h* pa = (const v8h*)(sA + m * HIDDEN + ks * KSLAB + half * 16);
      v8h lo = pa[0];
      v8h hi = pa[1];
      v16h afrag;
#pragma unroll
      for (int e = 0; e < 8; ++e) { afrag[e] = lo[e]; afrag[e + 8] = hi[e]; }
      acc[mt][0] = __builtin_amdgcn_wmma_f32_16x16x32_f16(
          false, afrag, false, bfrag[0], (short)0, acc[mt][0], false, false);
      acc[mt][1] = __builtin_amdgcn_wmma_f32_16x16x32_f16(
          false, afrag, false, bfrag[1], (short)0, acc[mt][1], false, false);
    }
  }
}

__launch_bounds__(THREADS, 1)
__global__ void vortex_fused(const float* __restrict__ nf,
                             const float* __restrict__ curv,
                             const _Float16* __restrict__ w1,
                             const float* __restrict__ b1,
                             const _Float16* __restrict__ w2,
                             const float* __restrict__ b2,
                             const float* __restrict__ mw,
                             float* __restrict__ out) {
  extern __shared__ char smem[];
  float*    sH = (float*)(smem + SH_OFF);
  _Float16* sA = (_Float16*)(smem + SA_OFF);
  _Float16* sB = (_Float16*)(smem + SB_OFF);

  const int tid   = threadIdx.x;
  const int lane  = tid & 31;
  const int wave  = tid >> 5;
  const int l16   = lane & 15;
  const int mhalf = (lane >> 4) * 8;
  const long long rowBase = (long long)blockIdx.x * ROWS;

  // ---- stage node_features tile (rows are contiguous: row = block*72 + r) ----
  {
    const float4* src = (const float4*)(nf + rowBase * HIDDEN);
    v4h* dst = (v4h*)sA;
    for (int i = tid; i < ROWS * HIDDEN / 4; i += THREADS) {
      float4 v = src[i];
      v4h h;
      h[0] = (_Float16)v.x; h[1] = (_Float16)v.y;
      h[2] = (_Float16)v.z; h[3] = (_Float16)v.w;
      dst[i] = h;
    }
    const v4h z = {(_Float16)0.f, (_Float16)0.f, (_Float16)0.f, (_Float16)0.f};
    for (int i = ROWS * HIDDEN / 4 + tid; i < ROWS_PAD * HIDDEN / 4; i += THREADS)
      dst[i] = z;  // zero pad rows 72..79
  }
  __syncthreads();

  v8f acc[5][2];

  // ---- GEMM1: x_proj = A @ to_w^T + b1  -> sH (f32) ----
  run_gemm(sA, sB, w1, tid, acc);
#pragma unroll
  for (int mt = 0; mt < 5; ++mt)
#pragma unroll
    for (int j = 0; j < 2; ++j) {
      const int n = wave * 32 + j * 16 + l16;
      const float bias = b1[n];
#pragma unroll
      for (int r = 0; r < 8; ++r) {
        const int m = mt * 16 + mhalf + r;
        sH[m * HIDDEN + n] = acc[mt][j][r] + bias;
      }
    }
  __syncthreads();

  // ---- h = tanh(||x||) * x / (||x|| + eps), in place in sH ----
#pragma unroll
  for (int rr = 0; rr < 5; ++rr) {
    const int m = wave * 5 + rr;  // 16 waves * 5 = 80 rows
    float vals[16];
    float ss = 0.f;
#pragma unroll
    for (int t = 0; t < 16; ++t) {
      vals[t] = sH[m * HIDDEN + lane + 32 * t];
      ss += vals[t] * vals[t];
    }
    ss = wave_sum(ss);
    const float nrm = sqrtf(ss);
    const float sc  = tanhf(nrm) / (nrm + EPSF);
#pragma unroll
    for (int t = 0; t < 16; ++t) sH[m * HIDDEN + lane + 32 * t] = vals[t] * sc;
  }
  __syncthreads();

  // ---- Mobius message passing: one chain (batch b, node i) per wave ----
  {
    const float c = fabsf(curv[0]);
    for (int chain = wave; chain < ROWS; chain += NWAVES) {
      const int b = chain / NUM_NODES;
      const int i = chain % NUM_NODES;
      float r[16];
#pragma unroll
      for (int t = 0; t < 16; ++t) r[t] = sH[chain * HIDDEN + lane + 32 * t];
      const int e1 = g_off[i + 1];
      for (int e = g_off[i]; e < e1; ++e) {
        const int j = g_nbr[e];
        const float* wp = mw + ((size_t)i * NUM_NODES + j) * HIDDEN;
        const int rowj = b * NUM_NODES + j;
        float x[16], y[16], tv[16], rn[16];
#pragma unroll
        for (int t = 0; t < 16; ++t) {
          x[t] = sH[rowj * HIDDEN + lane + 32 * t];
          y[t] = wp[lane + 32 * t];
        }
        mobius_add(x, y, c, tv);   // t = mobius(h_j, w_ij)
        mobius_add(r, tv, c, rn);  // r = mobius(r, t)
#pragma unroll
        for (int t = 0; t < 16; ++t) r[t] = rn[t];
      }
#pragma unroll
      for (int t = 0; t < 16; ++t)
        sA[chain * HIDDEN + lane + 32 * t] = (_Float16)r[t];  // result (f16) for GEMM2
    }
  }
  __syncthreads();

  // ---- GEMM2: out = result @ from_w^T + b2 ----
  run_gemm(sA, sB, w2, tid, acc);
#pragma unroll
  for (int mt = 0; mt < 5; ++mt)
#pragma unroll
    for (int j = 0; j < 2; ++j) {
      const int n = wave * 32 + j * 16 + l16;
      const float bias = b2[n];
#pragma unroll
      for (int r = 0; r < 8; ++r) {
        const int m = mt * 16 + mhalf + r;
        if (m < ROWS)
          out[(rowBase + m) * HIDDEN + n] = acc[mt][j][r] + bias;
      }
    }
}

extern "C" void kernel_launch(void* const* d_in, const int* in_sizes, int n_in,
                              void* d_out, int out_size, void* d_ws, size_t ws_size,
                              hipStream_t stream) {
  (void)in_sizes; (void)n_in; (void)out_size; (void)ws_size;
  const float* nf   = (const float*)d_in[0];  // (8192, 9, 512)
  const float* curv = (const float*)d_in[1];  // scalar
  const float* to_w = (const float*)d_in[2];  // (512, 512)
  const float* to_b = (const float*)d_in[3];  // (512,)
  const float* fr_w = (const float*)d_in[4];  // (512, 512)
  const float* fr_b = (const float*)d_in[5];  // (512,)
  const float* mw   = (const float*)d_in[6];  // (9, 9, 512)
  float* out = (float*)d_out;

  // workspace: f16 copies of both weight matrices (1 MB total)
  _Float16* w1h = (_Float16*)d_ws;
  _Float16* w2h = w1h + HIDDEN * HIDDEN;

  cvt_weights<<<(2 * HIDDEN * HIDDEN + 255) / 256, 256, 0, stream>>>(to_w, fr_w, w1h, w2h);

  hipFuncSetAttribute((const void*)vortex_fused,
                      hipFuncAttributeMaxDynamicSharedMemorySize, SMEM_BYTES);
  vortex_fused<<<BATCH / BT, THREADS, SMEM_BYTES, stream>>>(
      nf, curv, w1h, to_b, w2h, fr_b, mw, out);
}
